// SECOND_NECK_BASE_34411277975638
// MI455X (gfx1250) — compile-verified, matrix operand order checked by reading, not run on hardware
//
#include <hip/hip_runtime.h>
#include <hip/hip_bf16.h>

// ---------------------------------------------------------------------------
// Types
// ---------------------------------------------------------------------------
typedef __attribute__((ext_vector_type(16))) _Float16 v16h;
typedef __attribute__((ext_vector_type(2)))  __fp16   v2fp16;  // cvt_pkrtz result type
typedef __attribute__((ext_vector_type(8)))  float    v8f;
typedef __attribute__((ext_vector_type(4)))  float    f32x4;
typedef unsigned int u32;
typedef __attribute__((ext_vector_type(2))) u32 u32x2;
typedef __attribute__((ext_vector_type(4))) u32 u32x4;

#define DD 8
#define HH 160
#define WW 160
#define BB 2
#define GD (DD + 2)
#define GH (HH + 2)
#define GW (WW + 2)
#define COUT_FINAL 128

// ---------------------------------------------------------------------------
// Rulebook construction
// ---------------------------------------------------------------------------
__global__ void fill_grid_kernel(int* g, int total, int n) {
    int i = blockIdx.x * blockDim.x + threadIdx.x;
    int stride = gridDim.x * blockDim.x;
    for (; i < total; i += stride) g[i] = n;
}

__global__ void scatter_grid_kernel(const int* __restrict__ coors, int* __restrict__ g, int n) {
    int i = blockIdx.x * blockDim.x + threadIdx.x;
    if (i >= n) return;
    int b = coors[4 * i + 0];
    int z = coors[4 * i + 1] + 1;
    int y = coors[4 * i + 2] + 1;
    int x = coors[4 * i + 3] + 1;
    g[((b * GD + z) * GH + y) * GW + x] = i;
}

__global__ void build_nbr_kernel(const int* __restrict__ coors, const int* __restrict__ g,
                                 int* __restrict__ nbr, int n) {
    int i = blockIdx.x * blockDim.x + threadIdx.x;
    if (i >= n) return;
    int b = coors[4 * i + 0];
    int z = coors[4 * i + 1] + 1;
    int y = coors[4 * i + 2] + 1;
    int x = coors[4 * i + 3] + 1;
    int off = 0;
    for (int dz = -1; dz <= 1; ++dz)
        for (int dy = -1; dy <= 1; ++dy)
            for (int dx = -1; dx <= 1; ++dx) {
                nbr[off * n + i] = g[((b * GD + z + dz) * GH + y + dy) * GW + x + dx];
                ++off;
            }
}

// ---------------------------------------------------------------------------
// Weight packing: wt[k][o][ka] f16, ka = c*4 + {0: base, 1..3: spline g}
// Transposed layout => B fragments are contiguous K-runs of 8 halves in LDS.
// ---------------------------------------------------------------------------
__global__ void pack_w_kernel(const float* __restrict__ wb, const float* __restrict__ ws,
                              _Float16* __restrict__ wt, int CIN, int COUT, int total) {
    int e = blockIdx.x * blockDim.x + threadIdx.x;
    int stride = gridDim.x * blockDim.x;
    int KA = 4 * CIN;
    for (; e < total; e += stride) {
        int ka = e % KA;
        int o  = (e / KA) % COUT;
        int k  = e / (KA * COUT);
        int c  = ka >> 2;
        int r  = ka & 3;
        float v = (r == 0) ? wb[(k * CIN + c) * COUT + o]
                           : ws[((k * CIN + c) * 3 + (r - 1)) * COUT + o];
        wt[e] = (_Float16)v;
    }
}

// ---------------------------------------------------------------------------
// Fragment load from LDS: 16 halves/lane = 2 x ds_load_b128
// Layout (ISA 7.12.2, 16-bit 16x32): lane<16 -> K 0-7 & 16-23; lane>=16 -> K 8-15 & 24-31
// ---------------------------------------------------------------------------
__device__ inline v16h lds_frag(const _Float16* p) {
    union { v16h v; u32x4 q[2]; } u;
    u.q[0] = *(const u32x4*)(p);
    u.q[1] = *(const u32x4*)(p + 16);
    return u.v;
}

// ---------------------------------------------------------------------------
// Fused KAN conv block (pre-BN accumulation), f16 WMMA, f32 accumulators.
// 256 threads = 8 waves; workgroup covers 128 rows; wave w owns rows [w*16, w*16+16).
// K processed in 64-half chunks (16 channels) staged in LDS.
// A-tile build: thread t handles row t/2, channels [(t&1)*8, +8) of the chunk:
//   one nbr lookup, two global_load_b128 gathers, packed f16 conversion.
// ---------------------------------------------------------------------------
template <int CIN, int COUT>
__global__ __launch_bounds__(256)
void kan_conv_kernel(const float* __restrict__ feat, const int* __restrict__ nbr,
                     const _Float16* __restrict__ wt, float* __restrict__ outPre, int n) {
    constexpr int KA  = 4 * CIN;   // augmented K
    constexpr int KC  = 64;        // halves per K-chunk (16 channels)
    constexpr int NKC = KA / KC;   // K-chunks
    constexpr int NT  = COUT / 16; // output column tiles per wave

    __shared__ _Float16 As[128 * KC];   // 16 KB activation chunk
    __shared__ _Float16 Ws[COUT * KC];  // <= 16 KB weight chunk, [o][k] layout

    const int tid  = threadIdx.x;
    const int lane = tid & 31;
    const int wave = tid >> 5;
    const int row0 = blockIdx.x * 128;

    const int arow = tid >> 1;          // row in tile handled by this thread
    const int acb  = (tid & 1) * 8;     // channel base within chunk

    v8f acc[NT];
#pragma unroll
    for (int t = 0; t < NT; ++t) acc[t] = (v8f){};

    const int m  = lane & 15;
    const int kb = (lane < 16) ? 0 : 8;

    for (int k = 0; k < 27; ++k) {
        const int grow = row0 + arow;
        const int idx  = (grow < n) ? nbr[k * n + grow] : n;
        const bool vld = (idx < n);

        for (int kc = 0; kc < NKC; ++kc) {
            __syncthreads();  // protect LDS from previous iteration's readers

            // ---- build augmented activation chunk: 128 rows x 16 channels x 4 feats
            float xv[8];
            if (vld) {
                const f32x4* fp4 = (const f32x4*)&feat[idx * CIN + kc * 16 + acb];
                f32x4 a = fp4[0], b = fp4[1];
                xv[0] = a.x; xv[1] = a.y; xv[2] = a.z; xv[3] = a.w;
                xv[4] = b.x; xv[5] = b.y; xv[6] = b.z; xv[7] = b.w;
            } else {
#pragma unroll
                for (int j = 0; j < 8; ++j) xv[j] = 0.f;
            }
#pragma unroll
            for (int j = 0; j < 8; ++j) {
                float x = xv[j];
                // silu: x * sigmoid(x) via fast reciprocal (no IEEE divide)
                float s = x * __builtin_amdgcn_rcpf(1.f + __expf(-x));
                // RBF basis, centers {-1,0,1}, inv_h = 1
                float xm = x + 1.f, xp = x - 1.f;
                union { v2fp16 h; u32 u; } c0, c1;
                c0.h = __builtin_amdgcn_cvt_pkrtz(s, __expf(-xm * xm));
                c1.h = __builtin_amdgcn_cvt_pkrtz(__expf(-x * x), __expf(-xp * xp));
                u32x2 st; st.x = c0.u; st.y = c1.u;
                *(u32x2*)&As[arow * KC + (acb + j) * 4] = st;
            }

            // ---- stage weight chunk [COUT][KC] from wt[k] (transposed layout, contiguous)
            constexpr int Q = COUT * KC / 8;    // 16-byte transfers
            for (int q = tid; q < Q; q += 256) {
                int o  = q >> 3;                // 8 x uint4 per 64-half row
                int c8 = q & 7;
                *(u32x4*)&Ws[o * KC + c8 * 8] =
                    *(const u32x4*)&wt[(k * COUT + o) * KA + kc * KC + c8 * 8];
            }
            __syncthreads();

            // ---- WMMA over the two 32-K tiles of this chunk
#pragma unroll
            for (int kt = 0; kt < 2; ++kt) {
                v16h a = lds_frag(&As[(wave * 16 + m) * KC + kt * 32 + kb]);
#pragma unroll
                for (int nt = 0; nt < NT; ++nt) {
                    v16h b = lds_frag(&Ws[(nt * 16 + m) * KC + kt * 32 + kb]);
                    acc[nt] = __builtin_amdgcn_wmma_f32_16x16x32_f16(
                        false, a, false, b, (short)0, acc[nt], false, false);
                }
            }
        }
    }

    // ---- store pre-BN accumulators (C/D layout: VGPR r -> M=r / M=r+8)
    const int col0  = lane & 15;
    const int mbase = (lane < 16) ? 0 : 8;
#pragma unroll
    for (int nt = 0; nt < NT; ++nt) {
#pragma unroll
        for (int r = 0; r < 8; ++r) {
            int grow2 = row0 + wave * 16 + mbase + r;
            if (grow2 < n) outPre[grow2 * COUT + nt * 16 + col0] = acc[nt][r];
        }
    }
}

// ---------------------------------------------------------------------------
// Train-mode BatchNorm: per-channel batch stats folded into affine, then ReLU
// ---------------------------------------------------------------------------
template <int COUT>
__global__ __launch_bounds__(256)
void bn_stats_kernel(const float* __restrict__ pre, float* __restrict__ stats,
                     const float* __restrict__ gm, const float* __restrict__ bt, int n) {
    __shared__ float ssum[256];
    __shared__ float ssq[256];
    int co = blockIdx.x;
    float s = 0.f, q = 0.f;
    for (int i = threadIdx.x; i < n; i += 256) {
        float v = pre[i * COUT + co];
        s += v;
        q += v * v;
    }
    ssum[threadIdx.x] = s;
    ssq[threadIdx.x]  = q;
    __syncthreads();
    for (int d = 128; d > 0; d >>= 1) {
        if (threadIdx.x < d) {
            ssum[threadIdx.x] += ssum[threadIdx.x + d];
            ssq[threadIdx.x]  += ssq[threadIdx.x + d];
        }
        __syncthreads();
    }
    if (threadIdx.x == 0) {
        float inv = 1.f / (float)n;
        float mu  = ssum[0] * inv;
        float var = ssq[0] * inv - mu * mu;
        float a   = gm[co] * rsqrtf(var + 1e-3f);
        stats[2 * co + 0] = a;
        stats[2 * co + 1] = bt[co] - mu * a;
    }
}

template <int COUT>
__global__ void bn_norm_kernel(const float* __restrict__ pre, const float* __restrict__ stats,
                               float* __restrict__ out, int n) {
    int i = blockIdx.x * blockDim.x + threadIdx.x;
    if (i >= n * COUT) return;
    int co = i % COUT;
    float v = pre[i] * stats[2 * co] + stats[2 * co + 1];
    out[i] = v > 0.f ? v : 0.f;
}

// ---------------------------------------------------------------------------
// Output: zero dense tensor, scatter final features to [B, C*D, H, W]
// ---------------------------------------------------------------------------
__global__ void zero_f32_kernel(float* p, long total) {
    long i = (long)blockIdx.x * blockDim.x + threadIdx.x;
    long stride = (long)gridDim.x * blockDim.x;
    for (; i < total; i += stride) p[i] = 0.f;
}

__global__ void scatter_out_kernel(const float* __restrict__ feat, const int* __restrict__ coors,
                                   float* __restrict__ out, int n) {
    int t = blockIdx.x * blockDim.x + threadIdx.x;
    if (t >= n * COUT_FINAL) return;
    int i = t >> 7;
    int c = t & (COUT_FINAL - 1);
    int b = coors[4 * i + 0];
    int z = coors[4 * i + 1];
    int y = coors[4 * i + 2];
    int x = coors[4 * i + 3];
    long off = (((long)(b * COUT_FINAL + c) * DD + z) * HH + y) * WW + x;
    out[off] = feat[i * COUT_FINAL + c];
}

// ---------------------------------------------------------------------------
// Launch
// ---------------------------------------------------------------------------
extern "C" void kernel_launch(void* const* d_in, const int* in_sizes, int n_in,
                              void* d_out, int out_size, void* d_ws, size_t ws_size,
                              hipStream_t stream) {
    const int n = in_sizes[0] / 16;  // N_VOX

    const float* vf    = (const float*)d_in[0];
    const int*   coors = (const int*)d_in[1];
    // blocks: [3..6]=in, [7..10]=e1, [11..14]=e2, [15..18]=e3, [19..22]=out
    const float* wb[5]; const float* ws[5]; const float* gm[5]; const float* bt[5];
    for (int bkl = 0; bkl < 5; ++bkl) {
        wb[bkl] = (const float*)d_in[3 + 4 * bkl];
        ws[bkl] = (const float*)d_in[4 + 4 * bkl];
        gm[bkl] = (const float*)d_in[5 + 4 * bkl];
        bt[bkl] = (const float*)d_in[6 + 4 * bkl];
    }
    const int CI[5] = {16, 16, 32, 64, 64};
    const int CO[5] = {16, 32, 64, 64, 128};

    // ---- workspace carve
    char* p = (char*)d_ws;
    auto alloc = [&](size_t bytes) -> char* {
        char* r = p;
        p += (bytes + 255) & ~(size_t)255;
        return r;
    };
    const int GSZ = BB * GD * GH * GW;
    int*       vgrid = (int*)alloc((size_t)GSZ * 4);
    int*       nbr   = (int*)alloc((size_t)27 * n * 4);
    _Float16*  wt[5];
    for (int bkl = 0; bkl < 5; ++bkl)
        wt[bkl] = (_Float16*)alloc((size_t)27 * 4 * CI[bkl] * CO[bkl] * 2);
    float* bufA  = (float*)alloc((size_t)n * 128 * 4);
    float* bufB  = (float*)alloc((size_t)n * 128 * 4);
    float* preB  = (float*)alloc((size_t)n * 128 * 4);
    float* stats = (float*)alloc(256 * 4);

    // ---- rulebook
    fill_grid_kernel<<<1024, 256, 0, stream>>>(vgrid, GSZ, n);
    scatter_grid_kernel<<<(n + 255) / 256, 256, 0, stream>>>(coors, vgrid, n);
    build_nbr_kernel<<<(n + 255) / 256, 256, 0, stream>>>(coors, vgrid, nbr, n);

    // ---- pack weights (f16, transposed [k][o][4*Cin])
    for (int bkl = 0; bkl < 5; ++bkl) {
        int total = 27 * 4 * CI[bkl] * CO[bkl];
        pack_w_kernel<<<(total + 255) / 256, 256, 0, stream>>>(
            wb[bkl], ws[bkl], wt[bkl], CI[bkl], CO[bkl], total);
    }

    const int gconv = (n + 127) / 128;

    // ---- block 'in' : vf(16) -> bufA(16)
    kan_conv_kernel<16, 16><<<gconv, 256, 0, stream>>>(vf, nbr, wt[0], preB, n);
    bn_stats_kernel<16><<<16, 256, 0, stream>>>(preB, stats, gm[0], bt[0], n);
    bn_norm_kernel<16><<<(n * 16 + 255) / 256, 256, 0, stream>>>(preB, stats, bufA, n);

    // ---- block 'e1' : bufA(16) -> bufB(32)
    kan_conv_kernel<16, 32><<<gconv, 256, 0, stream>>>(bufA, nbr, wt[1], preB, n);
    bn_stats_kernel<32><<<32, 256, 0, stream>>>(preB, stats, gm[1], bt[1], n);
    bn_norm_kernel<32><<<(n * 32 + 255) / 256, 256, 0, stream>>>(preB, stats, bufB, n);

    // ---- block 'e2' : bufB(32) -> bufA(64)
    kan_conv_kernel<32, 64><<<gconv, 256, 0, stream>>>(bufB, nbr, wt[2], preB, n);
    bn_stats_kernel<64><<<64, 256, 0, stream>>>(preB, stats, gm[2], bt[2], n);
    bn_norm_kernel<64><<<(n * 64 + 255) / 256, 256, 0, stream>>>(preB, stats, bufA, n);

    // ---- block 'e3' : bufA(64) -> bufB(64)
    kan_conv_kernel<64, 64><<<gconv, 256, 0, stream>>>(bufA, nbr, wt[3], preB, n);
    bn_stats_kernel<64><<<64, 256, 0, stream>>>(preB, stats, gm[3], bt[3], n);
    bn_norm_kernel<64><<<(n * 64 + 255) / 256, 256, 0, stream>>>(preB, stats, bufB, n);

    // ---- block 'out' : bufB(64) -> bufA(128)
    kan_conv_kernel<64, 128><<<gconv, 256, 0, stream>>>(bufB, nbr, wt[4], preB, n);
    bn_stats_kernel<128><<<128, 256, 0, stream>>>(preB, stats, gm[4], bt[4], n);
    bn_norm_kernel<128><<<(n * 128 + 255) / 256, 256, 0, stream>>>(preB, stats, bufA, n);

    // ---- dense output
    zero_f32_kernel<<<4096, 256, 0, stream>>>((float*)d_out, (long)out_size);
    scatter_out_kernel<<<(n * COUT_FINAL + 255) / 256, 256, 0, stream>>>(
        bufA, coors, (float*)d_out, n);
}
